// SCNNModel_45870250721488
// MI455X (gfx1250) — matrix operand order repaired and checked by reading, not compiled, not run on hardware
//
#include <hip/hip_runtime.h>

// ---------------------------------------------------------------------------
// CDNA5 (gfx1250) fused spherical-CNN forward pass.
// All matmuls via v_wmma_f32_16x16x32_bf16 (bf16 operands, fp32 accumulate).
// Activations + all GEMM operands resident in LDS:
//   2 x 128KB activation ping-pong, 21KB SH transforms, 24KB stage that
//   time-shares per-layer weights (sconv phase) and sig scratch (nonlin phase).
// ---------------------------------------------------------------------------

typedef __bf16 bf16;
typedef __attribute__((ext_vector_type(16))) __bf16 v16bf;
typedef __attribute__((ext_vector_type(8)))  float  v8f;

#define NB      16                      // batch tile per workgroup
#define CPAD    64                      // channel padding (max channels)
#define KPAD    64                      // SH-coefficient padding (45 -> 64)
#define SIGP    96                      // direction padding (90 -> 96)
#define NWAVES  8                       // 256 threads, wave32
#define BUF_ELEMS   (NB * KPAD * CPAD)  // 65536 bf16 = 128 KB
#define BISFT_ELEMS (SIGP * KPAD)       // 6144 bf16
#define BSFT_ELEMS  (48 * SIGP)         // 4608 bf16
#define STAGE_ELEMS (NWAVES * NB * SIGP)// 12288 bf16 = 24 KB (>= max W layer)

// ---- WMMA wrapper -----------------------------------------------------------
__device__ __forceinline__ v8f wmma_bf16(v16bf a, v16bf b, v8f c) {
  // 8 args: (neg_a, A, neg_b, B, c_mod, C, reuse_a, reuse_b)
  return __builtin_amdgcn_wmma_f32_16x16x32_bf16(false, a, false, b, (short)0, c,
                                                 false, false);
}

// ---- A operand (16x32 bf16): lane l -> row m=l&15; half=l>>4 selects K-phase.
// a[0..7]  = K = half*16 + i       (VGPR0..3)
// a[8..15] = K = half*16 + 8 + i   (VGPR4..7)
__device__ __forceinline__ v16bf load_A(const bf16* p, int rs, int ks, int kbase) {
  const int l    = threadIdx.x & 31;
  const int m    = l & 15;
  const int half = l >> 4;
  const bf16* q = p + m * rs + (kbase + half * 16) * ks;
  v16bf a;
#pragma unroll
  for (int i = 0; i < 8; ++i) a[i] = q[i * ks];
  q += 8 * ks;
#pragma unroll
  for (int i = 0; i < 8; ++i) a[8 + i] = q[i * ks];
  return a;
}

// ---- B operand (32x16 bf16): lane l -> column n=l&15; khalf=l>>4.
// b[i] = B[K = khalf*16 + i][n]; storage is [N][Kpad] so K is contiguous.
__device__ __forceinline__ v16bf load_B(const bf16* p, int colstride, int kbase) {
  const int l     = threadIdx.x & 31;
  const int n     = l & 15;
  const int khalf = l >> 4;
  const bf16* q = p + n * colstride + kbase + khalf * 16;
  v16bf b;
#pragma unroll
  for (int i = 0; i < 16; ++i) b[i] = q[i];
  return b;
}

// ---- D (16x16 f32): lane l -> column n=l&15; VGPR r -> row M = r + 8*(l>>4).
__device__ __forceinline__ void store_D_scaled(bf16* p, int rs, int ns, v8f d,
                                               float scl) {
  const int l    = threadIdx.x & 31;
  const int n    = l & 15;
  const int half = l >> 4;
#pragma unroll
  for (int r = 0; r < 8; ++r) {
    const int m = r + 8 * half;
    p[m * rs + n * ns] = (bf16)(d[r] * scl);
  }
}

__device__ __forceinline__ void store_D_relu(bf16* p, int rs, int ns, v8f d) {
  const int l    = threadIdx.x & 31;
  const int n    = l & 15;
  const int half = l >> 4;
#pragma unroll
  for (int r = 0; r < 8; ++r) {
    const int m = r + 8 * half;
    p[m * rs + n * ns] = (bf16)fmaxf(d[r], 0.0f);
  }
}

// ---------------------------------------------------------------------------
// Weight/transform packing kernel: fp32 -> zero-padded bf16, B-operand layout
// ([N][Kpad], K contiguous per column).
// ---------------------------------------------------------------------------
__device__ __forceinline__ void pack_w(bf16* __restrict__ P,
                                       const float* __restrict__ w, int cout,
                                       int cin, int Np, int Kp, int idx0,
                                       int stride) {
  const int tot = 5 * Np * Kp;
  for (int i = idx0; i < tot; i += stride) {
    const int g = i / (Np * Kp);
    const int r = i % (Np * Kp);
    const int o = r / Kp;
    const int c = r % Kp;
    P[i] = (bf16)((o < cout && c < cin) ? w[(o * cin + c) * 5 + g] : 0.0f);
  }
}

__global__ __launch_bounds__(256) void scnn_pack(
    const float* __restrict__ sft, const float* __restrict__ isft,
    const float* __restrict__ w1, const float* __restrict__ w2,
    const float* __restrict__ w3, const float* __restrict__ w4,
    const float* __restrict__ w5, const float* __restrict__ w6,
    bf16* __restrict__ Bisft, bf16* __restrict__ Bsft, bf16* __restrict__ P0,
    bf16* __restrict__ P1, bf16* __restrict__ P2, bf16* __restrict__ P3,
    bf16* __restrict__ P4, bf16* __restrict__ P5) {
  const int idx0   = blockIdx.x * 256 + threadIdx.x;
  const int stride = gridDim.x * 256;
  // isft^T padded: B for step1, K=k(64), N=p(96): Bisft[p][k]
  for (int i = idx0; i < BISFT_ELEMS; i += stride) {
    const int p = i >> 6, k = i & 63;
    Bisft[i] = (bf16)((p < 90 && k < 45) ? isft[p * 45 + k] : 0.0f);
  }
  // sft padded: B for step2, K=p(96), N=k(48): Bsft[k][p]
  for (int i = idx0; i < BSFT_ELEMS; i += stride) {
    const int k = i / SIGP, p = i % SIGP;
    Bsft[i] = (bf16)((k < 45 && p < 90) ? sft[k * 90 + p] : 0.0f);
  }
  pack_w(P0, w1, 16, 4,  16, 32, idx0, stride);
  pack_w(P1, w2, 32, 16, 32, 32, idx0, stride);
  pack_w(P2, w3, 64, 32, 64, 32, idx0, stride);
  pack_w(P3, w4, 32, 64, 32, 64, idx0, stride);
  pack_w(P4, w5, 16, 32, 16, 32, idx0, stride);
  pack_w(P5, w6, 4,  16, 16, 32, idx0, stride);
}

// ---------------------------------------------------------------------------
// Fused 6-layer kernel. One workgroup = 16 batch elements, activations in LDS
// as bf16 [b][k(64)][c(64)] (channel-contiguous).
// ---------------------------------------------------------------------------
__global__ __launch_bounds__(256) void scnn_fused(
    const float* __restrict__ x, float* __restrict__ out, int batch,
    const bf16* __restrict__ Bisft, const bf16* __restrict__ Bsft,
    const bf16* __restrict__ W0, const bf16* __restrict__ W1,
    const bf16* __restrict__ W2, const bf16* __restrict__ W3,
    const bf16* __restrict__ W4, const bf16* __restrict__ W5) {
  extern __shared__ bf16 smem[];
  bf16* bufA  = smem;                     // 65536
  bf16* bufB  = bufA + BUF_ELEMS;         // 65536
  bf16* lisft = bufB + BUF_ELEMS;         // 6144  (LDS copy of Bisft)
  bf16* lsft  = lisft + BISFT_ELEMS;      // 4608  (LDS copy of Bsft)
  bf16* stage = lsft + BSFT_ELEMS;        // 12288 (W cache / sig scratch)

  const int tid  = threadIdx.x;
  const int wave = tid >> 5;
  const int b0   = blockIdx.x * NB;

  // --- zero bufA; copy SH transforms into LDS; stage input -----------------
  {
    uint4* z = (uint4*)bufA;
#pragma unroll 4
    for (int i = tid; i < BUF_ELEMS / 8; i += 256) z[i] = make_uint4(0, 0, 0, 0);
    uint4*       d0 = (uint4*)lisft;
    const uint4* s0 = (const uint4*)Bisft;
    for (int i = tid; i < BISFT_ELEMS / 8; i += 256) d0[i] = s0[i];
    uint4*       d1 = (uint4*)lsft;
    const uint4* s1 = (const uint4*)Bsft;
    for (int i = tid; i < BSFT_ELEMS / 8; i += 256) d1[i] = s1[i];
  }
  __syncthreads();
  // input x[b][c][k] -> bufA[(b*64+k)*64+c]
  for (int i = tid; i < NB * 4 * 45; i += 256) {
    const int b = i / (4 * 45);
    const int r = i % (4 * 45);
    const int c = r / 45;
    const int k = r % 45;
    if (b0 + b < batch)
      bufA[(b * KPAD + k) * CPAD + c] =
          (bf16)x[((size_t)(b0 + b) * 4 + c) * 45 + k];
  }
  __syncthreads();

  const int   couts[6] = {16, 32, 64, 32, 16, 4};
  const int   Kps[6]   = {32, 32, 32, 64, 32, 32};
  const int   Nps[6]   = {16, 32, 64, 32, 16, 16};
  const bf16* Ws[6]    = {W0, W1, W2, W3, W4, W5};
  // scale[k] = sqrt(pi / (2l+1)), l = 2g
  const float SCL[5] = {1.7724539f, 0.7926655f, 0.5908179f, 0.4915900f,
                        0.4298830f};

  bf16* src = bufA;
  bf16* dst = bufB;

  for (int layer = 0; layer < 6; ++layer) {
    const int cout = couts[layer];
    const int Kp   = Kps[layer];
    const int Np   = Nps[layer];
    const int nt   = Np >> 4;   // N tiles of 16
    const int kt   = Kp >> 5;   // K steps of 32

    // --- zero destination buffer + stream this layer's weights into LDS ----
    {
      uint4* z = (uint4*)dst;
#pragma unroll 4
      for (int i = tid; i < BUF_ELEMS / 8; i += 256) z[i] = make_uint4(0, 0, 0, 0);
      uint4*       wd = (uint4*)stage;
      const uint4* ws4 = (const uint4*)Ws[layer];
      const int    wn  = (5 * Np * Kp) / 8;
      for (int i = tid; i < wn; i += 256) wd[i] = ws4[i];
    }
    __syncthreads();

    // --- sconv: per coefficient k, out[b,o] = sum_c x[b,c,k] * Wg[o,c] ------
    for (int k = wave; k < 45; k += NWAVES) {
      const int   g   = (k == 0) ? 0 : (k < 6) ? 1 : (k < 15) ? 2
                                     : (k < 28) ? 3 : 4;
      const float scl = SCL[g];
      for (int n = 0; n < nt; ++n) {
        v8f acc = {0.f, 0.f, 0.f, 0.f, 0.f, 0.f, 0.f, 0.f};
        for (int ks = 0; ks < kt; ++ks) {
          v16bf a = load_A(src + k * CPAD, KPAD * CPAD, 1, ks * 32);
          v16bf b = load_B(stage + (g * Np + n * 16) * Kp, Kp, ks * 32);
          acc = wmma_bf16(a, b, acc);
        }
        store_D_scaled(dst + k * CPAD + n * 16, KPAD * CPAD, 1, acc, scl);
      }
    }
    __syncthreads();

    // --- nonlinearity, in place on dst: one channel tile (16 batch rows) ----
    bf16* my = stage + wave * (NB * SIGP);   // stage re-used as sig scratch
    for (int c = wave; c < cout; c += NWAVES) {
      // step 1: sig(16 x 96) = X_rows(16 x 64) @ isft^T(64 x 96)
      v8f s[6];
#pragma unroll
      for (int n = 0; n < 6; ++n)
        s[n] = (v8f){0.f, 0.f, 0.f, 0.f, 0.f, 0.f, 0.f, 0.f};
#pragma unroll
      for (int ks = 0; ks < 2; ++ks) {
        v16bf a = load_A(dst + c, KPAD * CPAD, CPAD, ks * 32);
#pragma unroll
        for (int n = 0; n < 6; ++n)
          s[n] = wmma_bf16(a, load_B(lisft + n * 16 * KPAD, KPAD, ks * 32), s[n]);
      }
      // ReLU -> per-wave staging (re-forms the A operand layout)
#pragma unroll
      for (int n = 0; n < 6; ++n) store_D_relu(my + n * 16, SIGP, 1, s[n]);

      // step 2: Xnew(16 x 48) = relu(sig)(16 x 96) @ sft^T(96 x 48)
      v8f t[3];
#pragma unroll
      for (int n = 0; n < 3; ++n)
        t[n] = (v8f){0.f, 0.f, 0.f, 0.f, 0.f, 0.f, 0.f, 0.f};
#pragma unroll
      for (int ks = 0; ks < 3; ++ks) {
        v16bf a = load_A(my, SIGP, 1, ks * 32);
#pragma unroll
        for (int n = 0; n < 3; ++n)
          t[n] = wmma_bf16(a, load_B(lsft + n * 16 * SIGP, SIGP, ks * 32), t[n]);
      }
#pragma unroll
      for (int n = 0; n < 3; ++n)
        store_D_scaled(dst + (n * 16) * CPAD + c, KPAD * CPAD, CPAD, t[n], 1.0f);
    }
    __syncthreads();

    bf16* tmp = src; src = dst; dst = tmp;  // ping-pong
  }

  // --- emit final activations (layer 6 result sits in `src`) ---------------
  for (int i = tid; i < NB * 4 * 45; i += 256) {
    const int b = i / (4 * 45);
    const int r = i % (4 * 45);
    const int c = r / 45;
    const int k = r % 45;
    if (b0 + b < batch)
      out[((size_t)(b0 + b) * 4 + c) * 45 + k] =
          (float)src[(b * KPAD + k) * CPAD + c];
  }
}

// ---------------------------------------------------------------------------
// Host launcher
// ---------------------------------------------------------------------------
extern "C" void kernel_launch(void* const* d_in, const int* in_sizes, int n_in,
                              void* d_out, int out_size, void* d_ws,
                              size_t ws_size, hipStream_t stream) {
  const float* x    = (const float*)d_in[0];
  const float* sft  = (const float*)d_in[1];
  const float* isft = (const float*)d_in[2];
  const float* w1   = (const float*)d_in[3];
  const float* w2   = (const float*)d_in[4];
  const float* w3   = (const float*)d_in[5];
  const float* w4   = (const float*)d_in[6];
  const float* w5   = (const float*)d_in[7];
  const float* w6   = (const float*)d_in[8];
  float*       out  = (float*)d_out;

  const int batch = in_sizes[0] / (4 * 45);

  // workspace layout (bf16 elements)
  bf16* ws    = (bf16*)d_ws;
  bf16* Bisft = ws;                   // 96*64   = 6144
  bf16* Bsft  = Bisft + BISFT_ELEMS;  // 48*96   = 4608
  bf16* P0    = Bsft + BSFT_ELEMS;    // 5*16*32 = 2560
  bf16* P1    = P0 + 5 * 16 * 32;     // 5*32*32 = 5120
  bf16* P2    = P1 + 5 * 32 * 32;     // 5*64*32 = 10240
  bf16* P3    = P2 + 5 * 64 * 32;     // 5*32*64 = 10240
  bf16* P4    = P3 + 5 * 32 * 64;     // 5*16*32 = 2560
  bf16* P5    = P4 + 5 * 16 * 32;     // 5*16*32 = 2560

  scnn_pack<<<64, 256, 0, stream>>>(sft, isft, w1, w2, w3, w4, w5, w6, Bisft,
                                    Bsft, P0, P1, P2, P3, P4, P5);

  const size_t shmem =
      (size_t)(2 * BUF_ELEMS + BISFT_ELEMS + BSFT_ELEMS + STAGE_ELEMS) *
      sizeof(bf16);
  const int grid = (batch + NB - 1) / NB;
  scnn_fused<<<grid, 256, shmem, stream>>>(x, out, batch, Bisft, Bsft, P0, P1,
                                           P2, P3, P4, P5);
}
